// DeformableTransformerDecoderLayer_7267084665177
// MI455X (gfx1250) — matrix operand order repaired and checked by reading, not compiled, not run on hardware
//
#include <hip/hip_runtime.h>

// ---------------- types ----------------
typedef unsigned short u16;
typedef __attribute__((ext_vector_type(16))) __bf16 v16bf;
typedef __attribute__((ext_vector_type(8)))  float  v8f;
typedef __attribute__((ext_vector_type(4)))  unsigned int u32x4;

union Frag { u32x4 u[2]; v16bf v; };

// ---------------- problem constants ----------------
#define C_LQ   900
#define C_B    16
#define C_D    256
#define C_H    8
#define C_DH   32
#define C_NTOK (C_LQ * C_B)        // 14400
#define C_S    13294
#define C_MROW (C_S * C_B)         // 212704
#define C_DFF  1024
#define C_LQP  912                 // padded query length (57*16)
#define C_LKP  928                 // padded key length (29*32)

__device__ __forceinline__ u16 f2bf(float f) {
  unsigned u = __float_as_uint(f);
  unsigned r = u + 0x7FFFu + ((u >> 16) & 1u);   // RNE
  return (u16)(r >> 16);
}
__device__ __forceinline__ float bf2f(u16 h) {
  return __uint_as_float(((unsigned)h) << 16);
}
__device__ __forceinline__ v8f wmma_bf16(v16bf a, v16bf b, v8f c) {
  return __builtin_amdgcn_wmma_f32_16x16x32_bf16(false, a, false, b, (short)0, c, false, false);
}

// ---------------- elementwise / conversion kernels ----------------
__global__ __launch_bounds__(256) void prep_qk_kernel(
    const float* __restrict__ tgt, const float* __restrict__ pos,
    u16* __restrict__ qk_bf, u16* __restrict__ tgt_bf, int n) {
  int i = blockIdx.x * 256 + threadIdx.x;
  if (i >= n) return;
  float t = tgt[i];
  qk_bf[i]  = f2bf(t + pos[i]);
  tgt_bf[i] = f2bf(t);
}

__global__ __launch_bounds__(256) void pack_bf_kernel(
    const float* __restrict__ x, u16* __restrict__ y, long long n) {
  long long i = (long long)blockIdx.x * 256 + threadIdx.x;
  if (i >= n) return;
  y[i] = f2bf(x[i]);
}

// ---------------- TDM demonstration (guarded; clang-23 6-arg form) ----------------
__global__ __launch_bounds__(32) void tdm_warm_kernel(const u16* __restrict__ src) {
#if defined(__gfx1250__) && __has_builtin(__builtin_amdgcn_tensor_load_to_lds) && \
    __has_builtin(__builtin_amdgcn_s_wait_tensorcnt)
  typedef __attribute__((ext_vector_type(4))) unsigned int tdm4u;
  typedef __attribute__((ext_vector_type(8))) int tdm8i;
  typedef __attribute__((ext_vector_type(4))) int tdm4i;
  __shared__ __align__(16) u16 buf[64 * 32];
  unsigned lds_addr = (unsigned)(unsigned long long)(&buf[0]);
  unsigned long long ga = (unsigned long long)(size_t)src;
  // D# group0: count=1 | lds_addr | global_addr[56:0] | type=2
  tdm4u g0;
  g0.x = 1u;
  g0.y = lds_addr;
  g0.z = (unsigned)(ga & 0xffffffffu);
  g0.w = (unsigned)((ga >> 32) & 0x01ffffffu) | (2u << 30);
  // D# group1: 2-D tensor 64x32 (bf16), tile 64x32, stride 32 elements
  tdm8i g1;
  g1[0] = (1 << 16);                 // data_size = 1 (2 bytes), wg_mask = 0
  g1[1] = (32 & 0xffff) << 16;       // tensor_dim0[15:0] = 32
  g1[2] = ((64 & 0xffff) << 16);     // tensor_dim0 hi = 0, tensor_dim1[15:0] = 64
  g1[3] = (32 << 16);                // tensor_dim1 hi = 0, tile_dim0 = 32
  g1[4] = 64;                        // tile_dim1 = 64, tile_dim2 = 0
  g1[5] = 32;                        // tensor_dim0_stride = 32
  g1[6] = 0;
  g1[7] = 0;
  tdm4i gz4 = {0, 0, 0, 0};
  tdm8i gz8 = {0, 0, 0, 0, 0, 0, 0, 0};
  __builtin_amdgcn_tensor_load_to_lds(g0, g1, gz4, gz4, gz8, 0);
  __builtin_amdgcn_s_wait_tensorcnt((short)0);
  __syncthreads();
  volatile u16 sink = buf[threadIdx.x];
  (void)sink;
#else
  (void)src;
#endif
}

// ---------------- generic tiled WMMA GEMM ----------------
// C[M, N] = A[M, K](bf16, lda) * W^T  with W pre-packed bf16 [N][K] (ldw = K).
// B fragments come straight from Wp rows (contiguous k), no LDS transpose.
// outputs: Cf fp32 and/or Cb bf16 (ldc). grid = (N/64, ceil(M/64)), block 256.
__global__ __launch_bounds__(256) void gemm_bf16_kernel(
    const u16* __restrict__ A, int lda,
    const u16* __restrict__ Wp, int ldw,
    const float* __restrict__ bias,
    float* __restrict__ Cf, u16* __restrict__ Cb, int ldc,
    int M, int K, int relu) {
  __shared__ __align__(16) u16 As[64 * 32];

  int tid = threadIdx.x, w = tid >> 5, lane = tid & 31;
  int mBlock = blockIdx.y * 64, nBlock = blockIdx.x * 64;
  int m_off = (w & 3) * 16, n_off = (w >> 2) * 32;
  v8f acc0 = {}, acc1 = {};

  int arow = mBlock + (tid >> 2);
  int acg  = (tid & 3) * 8;
  bool aval = arow < M;
  const u16* ap = A + (size_t)arow * lda + acg;

  int mr = m_off + (lane & 15), kh = lane >> 4;
  const u16* wp0 = Wp + (size_t)(nBlock + n_off + (lane & 15)) * ldw + kh * 8;
  const u16* wp1 = wp0 + (size_t)16 * ldw;

  for (int kb = 0; kb < K; kb += 32) {
    u32x4 av = {};
    if (aval) {
      av = *(const u32x4*)(ap + kb);
      __builtin_prefetch(ap + kb + 64, 0, 1);   // global_prefetch_b8
    }
    __syncthreads();
    *(u32x4*)&As[(tid >> 2) * 32 + acg] = av;
    __syncthreads();

    Frag a, b0, b1;
    a.u[0] = *(const u32x4*)&As[mr * 32 + kh * 8];
    a.u[1] = *(const u32x4*)&As[mr * 32 + 16 + kh * 8];
    b0.u[0] = *(const u32x4*)(wp0 + kb);
    b0.u[1] = *(const u32x4*)(wp0 + kb + 16);
    b1.u[0] = *(const u32x4*)(wp1 + kb);
    b1.u[1] = *(const u32x4*)(wp1 + kb + 16);
    acc0 = wmma_bf16(a.v, b0.v, acc0);
    acc1 = wmma_bf16(a.v, b1.v, acc1);
  }

  int nc0 = nBlock + n_off + (lane & 15);
  float bv0 = bias ? bias[nc0] : 0.f;
  float bv1 = bias ? bias[nc0 + 16] : 0.f;
  #pragma unroll
  for (int r = 0; r < 8; r++) {
    int mr2 = mBlock + m_off + r + ((lane >= 16) ? 8 : 0);
    if (mr2 < M) {
      float v0 = acc0[r] + bv0;
      float v1 = acc1[r] + bv1;
      if (relu) { v0 = fmaxf(v0, 0.f); v1 = fmaxf(v1, 0.f); }
      if (Cf) {
        Cf[(size_t)mr2 * ldc + nc0]      = v0;
        Cf[(size_t)mr2 * ldc + nc0 + 16] = v1;
      }
      if (Cb) {
        Cb[(size_t)mr2 * ldc + nc0]      = f2bf(v0);
        Cb[(size_t)mr2 * ldc + nc0 + 16] = f2bf(v1);
      }
    }
  }
}

// ---------------- QKV split / repack ----------------
__global__ __launch_bounds__(256) void split_qkv_kernel(
    const float* __restrict__ qkv, u16* __restrict__ Qb, u16* __restrict__ Kb,
    u16* __restrict__ Vt) {
  int idx = blockIdx.x * 256 + threadIdx.x;
  const int TOT = C_B * C_H * C_LKP * C_DH;
  if (idx >= TOT) return;
  int d  = idx & 31;
  int r  = (idx >> 5) % C_LKP;
  int bh = idx / (C_LKP * C_DH);
  int b = bh >> 3, h = bh & 7;
  float q = 0.f, k = 0.f, v = 0.f;
  if (r < C_LQ) {
    const float* p = qkv + (size_t)(r * C_B + b) * 768 + h * C_DH + d;
    q = p[0]; k = p[256]; v = p[512];
  }
  if (r < C_LQP) {
    Qb[((size_t)bh * C_LQP + r) * C_DH + d] = f2bf(q);
    Kb[((size_t)bh * C_LQP + r) * C_DH + d] = f2bf(k);
  }
  Vt[((size_t)bh * C_DH + d) * C_LKP + r] = f2bf(v);
}

// ---------------- fused self-attention (per (qtile, h, b)) ----------------
// Dynamic LDS layout: S fp32[16][928] | P bf16[16][928] | Qs bf16[16][32] | Opart f32[8][512]
#define ATTN_SMEM (16 * C_LKP * 4 + 16 * C_LKP * 2 + 16 * 32 * 2 + 8 * 512 * 4)
__global__ __launch_bounds__(256) void attn_kernel(
    const u16* __restrict__ Qb, const u16* __restrict__ Kb,
    const u16* __restrict__ Vt, u16* __restrict__ O) {
  extern __shared__ __align__(16) char smem[];
  float* S  = (float*)smem;
  u16*   P  = (u16*)(smem + 16 * C_LKP * 4);
  u16*   Qs = (u16*)(smem + 16 * C_LKP * 4 + 16 * C_LKP * 2);
  float* Op = (float*)(smem + 16 * C_LKP * 4 + 16 * C_LKP * 2 + 1024);

  int qt = blockIdx.x, h = blockIdx.y, b = blockIdx.z;
  int bh = b * C_H + h;
  int tid = threadIdx.x, w = tid >> 5, lane = tid & 31;
  const u16* Qg = Qb + (size_t)bh * C_LQP * C_DH;
  const u16* Kg = Kb + (size_t)bh * C_LQP * C_DH;
  const u16* Vg = Vt + (size_t)bh * C_DH * C_LKP;

  if (tid < 64) {
    int r = tid >> 2, part = tid & 3;
    *(u32x4*)&Qs[r * 32 + part * 8] =
        *(const u32x4*)&Qg[(size_t)(qt * 16 + r) * 32 + part * 8];
  }
  __syncthreads();

  const float scale = 0.17677669529663687f;  // 1/sqrt(32)
  int m = lane & 15, kh = lane >> 4;
  for (int kt = w; kt < 57; kt += 8) {
    Frag a, bf;
    a.u[0] = *(const u32x4*)&Qs[m * 32 + kh * 8];
    a.u[1] = *(const u32x4*)&Qs[m * 32 + 16 + kh * 8];
    const u16* kp = Kg + (size_t)(kt * 16 + m) * 32 + kh * 8;
    bf.u[0] = *(const u32x4*)kp;
    bf.u[1] = *(const u32x4*)(kp + 16);
    v8f acc = {};
    acc = wmma_bf16(a.v, bf.v, acc);
    int nbase = kt * 16 + (lane & 15);
    #pragma unroll
    for (int r = 0; r < 8; r++) {
      int mm = r + ((lane >= 16) ? 8 : 0);
      float vsc = acc[r] * scale;
      if (nbase >= C_LQ) vsc = -1e30f;
      S[mm * C_LKP + nbase] = vsc;
    }
  }
  {
    int mm = tid >> 4, nn = C_LQP + (tid & 15);
    S[mm * C_LKP + nn] = -1e30f;
  }
  __syncthreads();

  for (int rr = 0; rr < 2; rr++) {
    int row = w * 2 + rr;
    float mx = -1e30f;
    for (int j = lane; j < C_LKP; j += 32) mx = fmaxf(mx, S[row * C_LKP + j]);
    for (int s = 16; s > 0; s >>= 1) mx = fmaxf(mx, __shfl_xor(mx, s));
    float sum = 0.f;
    for (int j = lane; j < C_LKP; j += 32) sum += __expf(S[row * C_LKP + j] - mx);
    for (int s = 16; s > 0; s >>= 1) sum += __shfl_xor(sum, s);
    float inv = 1.f / sum;
    for (int j = lane; j < C_LKP; j += 32)
      P[row * C_LKP + j] = f2bf(__expf(S[row * C_LKP + j] - mx) * inv);
  }
  __syncthreads();

  v8f acc0 = {}, acc1 = {};
  for (int s = w; s < 29; s += 8) {
    int kb = s * 32;
    Frag a, b0, b1;
    a.u[0] = *(const u32x4*)&P[m * C_LKP + kb + kh * 8];
    a.u[1] = *(const u32x4*)&P[m * C_LKP + kb + 16 + kh * 8];
    const u16* vp0 = Vg + (size_t)m * C_LKP + kb + kh * 8;
    const u16* vp1 = Vg + (size_t)(m + 16) * C_LKP + kb + kh * 8;
    b0.u[0] = *(const u32x4*)vp0; b0.u[1] = *(const u32x4*)(vp0 + 16);
    b1.u[0] = *(const u32x4*)vp1; b1.u[1] = *(const u32x4*)(vp1 + 16);
    acc0 = wmma_bf16(a.v, b0.v, acc0);
    acc1 = wmma_bf16(a.v, b1.v, acc1);
  }
  {
    float* op = Op + w * 512;
    int n = lane & 15;
    #pragma unroll
    for (int r = 0; r < 8; r++) {
      int mm = r + ((lane >= 16) ? 8 : 0);
      op[mm * 32 + n]      = acc0[r];
      op[mm * 32 + n + 16] = acc1[r];
    }
  }
  __syncthreads();
  for (int i = tid; i < 512; i += 256) {
    float sum = 0.f;
    #pragma unroll
    for (int ww = 0; ww < 8; ww++) sum += Op[ww * 512 + i];
    int mm = i >> 5, dd = i & 31;
    int qrow = qt * 16 + mm;
    if (qrow < C_LQ)
      O[((size_t)(qrow * C_B + b)) * C_D + h * C_DH + dd] = f2bf(sum);
  }
}

// ---------------- fused residual + LayerNorm ----------------
__global__ __launch_bounds__(256) void ln_kernel(
    const float* __restrict__ base, const float* __restrict__ delta,
    const float* __restrict__ g, const float* __restrict__ be,
    const float* __restrict__ pos, float* __restrict__ outF,
    u16* __restrict__ outB, int rows) {
  int row = blockIdx.x * 8 + (threadIdx.x >> 5);
  int lane = threadIdx.x & 31;
  if (row >= rows) return;
  size_t o = (size_t)row * C_D;
  float v[8];
  float s = 0.f;
  #pragma unroll
  for (int i = 0; i < 8; i++) { int d = lane + i * 32; v[i] = base[o + d] + delta[o + d]; s += v[i]; }
  for (int mm = 16; mm > 0; mm >>= 1) s += __shfl_xor(s, mm);
  float mean = s * (1.f / 256.f);
  float var = 0.f;
  #pragma unroll
  for (int i = 0; i < 8; i++) { float t = v[i] - mean; var += t * t; }
  for (int mm = 16; mm > 0; mm >>= 1) var += __shfl_xor(var, mm);
  float rstd = rsqrtf(var * (1.f / 256.f) + 1e-5f);
  #pragma unroll
  for (int i = 0; i < 8; i++) {
    int d = lane + i * 32;
    float y = (v[i] - mean) * rstd * g[d] + be[d];
    if (outF) outF[o + d] = y;
    if (outB) { float z = y + (pos ? pos[o + d] : 0.f); outB[o + d] = f2bf(z); }
  }
}

// ---------------- deformable bilinear gather ----------------
__global__ __launch_bounds__(256) void deform_gather_kernel(
    const float* __restrict__ off, const float* __restrict__ aw,
    const u16* __restrict__ val_bf, const float* __restrict__ refp,
    const int* __restrict__ shapes, const int* __restrict__ starts,
    u16* __restrict__ out_bf) {
  int q = blockIdx.x, b = blockIdx.y;
  int h = threadIdx.x >> 5, lane = threadIdx.x & 31;
  int row = q * C_B + b;

  float logit = (lane < 16) ? aw[(size_t)row * 128 + h * 16 + lane] : -1e30f;
  float mx = logit;
  for (int s = 16; s > 0; s >>= 1) mx = fmaxf(mx, __shfl_xor(mx, s));
  float e = (lane < 16) ? __expf(logit - mx) : 0.f;
  float ssum = e;
  for (int s = 16; s > 0; s >>= 1) ssum += __shfl_xor(ssum, s);
  float inv = 1.f / ssum;

  const float* offp = off + (size_t)row * C_D + h * 32;
  const float* rp = refp + (size_t)row * 16;
  float acc = 0.f;

  for (int l = 0; l < 4; l++) {
    int H_ = shapes[l * 2], W_ = shapes[l * 2 + 1];
    int s0 = starts[l];
    float rx = rp[l * 4 + 0], ry = rp[l * 4 + 1];
    float rw = rp[l * 4 + 2], rh = rp[l * 4 + 3];
    for (int p = 0; p < 4; p++) {
      float ox = offp[(l * 4 + p) * 2 + 0];
      float oy = offp[(l * 4 + p) * 2 + 1];
      float lx = rx + ox * 0.125f * rw;   // off/PTS * wh * 0.5
      float ly = ry + oy * 0.125f * rh;
      float x = lx * (float)W_ - 0.5f;
      float y = ly * (float)H_ - 0.5f;
      float x0 = floorf(x), y0 = floorf(y);
      float wx = x - x0, wy = y - y0;
      int xi = (int)x0, yi = (int)y0;
      float wgt = __shfl(e, l * 4 + p) * inv;
      #pragma unroll
      for (int c = 0; c < 4; c++) {
        int xc = xi + (c & 1), yc = yi + (c >> 1);
        float cw = ((c & 1) ? wx : 1.f - wx) * ((c >> 1) ? wy : 1.f - wy);
        bool valid = (xc >= 0) && (xc < W_) && (yc >= 0) && (yc < H_);
        int xq = min(max(xc, 0), W_ - 1);
        int yq = min(max(yc, 0), H_ - 1);
        float vv = bf2f(val_bf[((size_t)(s0 + yq * W_ + xq) * C_B + b) * C_D + h * 32 + lane]);
        acc += wgt * cw * (valid ? vv : 0.f);
      }
    }
  }
  out_bf[(size_t)row * C_D + h * 32 + lane] = f2bf(acc);
}

// ---------------- host ----------------
static inline int cdiv(long long a, long long b) { return (int)((a + b - 1) / b); }

extern "C" void kernel_launch(void* const* d_in, const int* in_sizes, int n_in,
                              void* d_out, int out_size, void* d_ws, size_t ws_size,
                              hipStream_t stream) {
  (void)in_sizes; (void)n_in; (void)out_size; (void)ws_size;
  const float* tgt   = (const float*)d_in[0];
  const float* pos   = (const float*)d_in[1];
  const float* refp  = (const float*)d_in[2];
  const float* mem   = (const float*)d_in[3];
  const int*   shapes= (const int*)d_in[4];
  const int*   starts= (const int*)d_in[5];
  const float* in_w  = (const float*)d_in[6];
  const float* in_b  = (const float*)d_in[7];
  const float* saw   = (const float*)d_in[8];
  const float* sab   = (const float*)d_in[9];
  const float* offw  = (const float*)d_in[10];
  const float* offb  = (const float*)d_in[11];
  const float* aww   = (const float*)d_in[12];
  const float* awb   = (const float*)d_in[13];
  const float* valw  = (const float*)d_in[14];
  const float* valb  = (const float*)d_in[15];
  const float* caw   = (const float*)d_in[16];
  const float* cab   = (const float*)d_in[17];
  const float* l1w   = (const float*)d_in[18];
  const float* l1b   = (const float*)d_in[19];
  const float* l2w   = (const float*)d_in[20];
  const float* l2b   = (const float*)d_in[21];
  const float* n1g   = (const float*)d_in[22];
  const float* n1b   = (const float*)d_in[23];
  const float* n2g   = (const float*)d_in[24];
  const float* n2b   = (const float*)d_in[25];
  const float* n3g   = (const float*)d_in[26];
  const float* n3b   = (const float*)d_in[27];
  float* out = (float*)d_out;

  char* base = (char*)d_ws;
  size_t o = 0;
  auto al = [&](size_t bytes) -> void* {
    void* p = base + o;
    o = (o + bytes + 255) & ~(size_t)255;
    return p;
  };
  // weights packed bf16, SAME layout as fp32 (row n of W = B-fragment column n)
  u16* w_qkv  = (u16*)al((size_t)768 * 256 * 2);
  u16* w_sa   = (u16*)al((size_t)256 * 256 * 2);
  u16* w_val  = (u16*)al((size_t)256 * 256 * 2);
  u16* w_offT = (u16*)al((size_t)256 * 256 * 2);
  u16* w_aw   = (u16*)al((size_t)128 * 256 * 2);
  u16* w_ca   = (u16*)al((size_t)256 * 256 * 2);
  u16* w_l1   = (u16*)al((size_t)1024 * 256 * 2);
  u16* w_l2   = (u16*)al((size_t)256 * 1024 * 2);
  u16*   qk_bf   = (u16*)al((size_t)C_NTOK * C_D * 2);
  u16*   tgt_bf  = (u16*)al((size_t)C_NTOK * C_D * 2);
  float* qkv     = (float*)al((size_t)C_NTOK * 768 * 4);
  u16*   Qbf     = (u16*)al((size_t)C_B * C_H * C_LQP * C_DH * 2);
  u16*   Kbf     = (u16*)al((size_t)C_B * C_H * C_LQP * C_DH * 2);
  u16*   Vtb     = (u16*)al((size_t)C_B * C_H * C_DH * C_LKP * 2);
  u16*   attn_bf = (u16*)al((size_t)C_NTOK * C_D * 2);
  float* sa_d    = (float*)al((size_t)C_NTOK * C_D * 4);
  float* tgt1    = (float*)al((size_t)C_NTOK * C_D * 4);
  u16*   q2_bf   = (u16*)al((size_t)C_NTOK * C_D * 2);
  u16*   mem_bf  = (u16*)al((size_t)C_MROW * C_D * 2);
  u16*   val_bf  = (u16*)al((size_t)C_MROW * C_D * 2);
  float* off_f   = (float*)al((size_t)C_NTOK * C_D * 4);
  float* aw_f    = (float*)al((size_t)C_NTOK * 128 * 4);
  u16*   ca_bf   = (u16*)al((size_t)C_NTOK * C_D * 2);
  float* ca_d    = (float*)al((size_t)C_NTOK * C_D * 4);
  float* tgt2    = (float*)al((size_t)C_NTOK * C_D * 4);
  u16*   tgt2_bf = (u16*)al((size_t)C_NTOK * C_D * 2);
  u16*   ffn1_bf = (u16*)al((size_t)C_NTOK * C_DFF * 2);
  float* ffn2    = (float*)al((size_t)C_NTOK * C_D * 4);

  (void)hipFuncSetAttribute((const void*)attn_kernel,
                            hipFuncAttributeMaxDynamicSharedMemorySize, ATTN_SMEM);

  // weights -> bf16 (no transpose needed: W[n][k] rows ARE the WMMA B columns)
  pack_bf_kernel<<<cdiv(768 * 256, 256), 256, 0, stream>>>(in_w, w_qkv, 768 * 256);
  pack_bf_kernel<<<cdiv(256 * 256, 256), 256, 0, stream>>>(saw, w_sa, 256 * 256);
  pack_bf_kernel<<<cdiv(256 * 256, 256), 256, 0, stream>>>(valw, w_val, 256 * 256);
  pack_bf_kernel<<<cdiv(256 * 256, 256), 256, 0, stream>>>(offw, w_offT, 256 * 256);
  pack_bf_kernel<<<cdiv(128 * 256, 256), 256, 0, stream>>>(aww, w_aw, 128 * 256);
  pack_bf_kernel<<<cdiv(256 * 256, 256), 256, 0, stream>>>(caw, w_ca, 256 * 256);
  pack_bf_kernel<<<cdiv(1024 * 256, 256), 256, 0, stream>>>(l1w, w_l1, 1024 * 256);
  pack_bf_kernel<<<cdiv(256 * 1024, 256), 256, 0, stream>>>(l2w, w_l2, 256 * 1024);

  // TDM demonstration / L2 warm of the QKV weights
  tdm_warm_kernel<<<1, 32, 0, stream>>>(w_qkv);

  // activations -> bf16
  prep_qk_kernel<<<cdiv((long long)C_NTOK * C_D, 256), 256, 0, stream>>>(
      tgt, pos, qk_bf, tgt_bf, C_NTOK * C_D);
  pack_bf_kernel<<<cdiv((long long)C_MROW * C_D, 256), 256, 0, stream>>>(
      mem, mem_bf, (long long)C_MROW * C_D);

  // ---- self-attention ----
  gemm_bf16_kernel<<<dim3(512 / 64, cdiv(C_NTOK, 64)), 256, 0, stream>>>(
      qk_bf, 256, w_qkv, 256, in_b, qkv, nullptr, 768, C_NTOK, 256, 0);       // Q,K
  gemm_bf16_kernel<<<dim3(256 / 64, cdiv(C_NTOK, 64)), 256, 0, stream>>>(
      tgt_bf, 256, w_qkv + (size_t)512 * 256, 256, in_b + 512, qkv + 512,
      nullptr, 768, C_NTOK, 256, 0);                                          // V
  split_qkv_kernel<<<cdiv((long long)C_B * C_H * C_LKP * C_DH, 256), 256, 0, stream>>>(
      qkv, Qbf, Kbf, Vtb);
  attn_kernel<<<dim3(C_LQP / 16, C_H, C_B), 256, ATTN_SMEM, stream>>>(
      Qbf, Kbf, Vtb, attn_bf);
  gemm_bf16_kernel<<<dim3(256 / 64, cdiv(C_NTOK, 64)), 256, 0, stream>>>(
      attn_bf, 256, w_sa, 256, sab, sa_d, nullptr, 256, C_NTOK, 256, 0);
  ln_kernel<<<cdiv(C_NTOK, 8), 256, 0, stream>>>(
      tgt, sa_d, n2g, n2b, pos, tgt1, q2_bf, C_NTOK);   // norm2; q2 = bf16(tgt1+pos)

  // ---- deformable cross-attention ----
  gemm_bf16_kernel<<<dim3(256 / 64, cdiv(C_MROW, 64)), 256, 0, stream>>>(
      mem_bf, 256, w_val, 256, valb, nullptr, val_bf, 256, C_MROW, 256, 0);   // value proj
  gemm_bf16_kernel<<<dim3(256 / 64, cdiv(C_NTOK, 64)), 256, 0, stream>>>(
      q2_bf, 256, w_offT, 256, offb, off_f, nullptr, 256, C_NTOK, 256, 0);    // offsets
  gemm_bf16_kernel<<<dim3(128 / 64, cdiv(C_NTOK, 64)), 256, 0, stream>>>(
      q2_bf, 256, w_aw, 256, awb, aw_f, nullptr, 128, C_NTOK, 256, 0);        // attn weights
  deform_gather_kernel<<<dim3(C_LQ, C_B), 256, 0, stream>>>(
      off_f, aw_f, val_bf, refp, shapes, starts, ca_bf);
  gemm_bf16_kernel<<<dim3(256 / 64, cdiv(C_NTOK, 64)), 256, 0, stream>>>(
      ca_bf, 256, w_ca, 256, cab, ca_d, nullptr, 256, C_NTOK, 256, 0);
  ln_kernel<<<cdiv(C_NTOK, 8), 256, 0, stream>>>(
      tgt1, ca_d, n1g, n1b, nullptr, tgt2, tgt2_bf, C_NTOK);  // norm1

  // ---- FFN ----
  gemm_bf16_kernel<<<dim3(1024 / 64, cdiv(C_NTOK, 64)), 256, 0, stream>>>(
      tgt2_bf, 256, w_l1, 256, l1b, nullptr, ffn1_bf, 1024, C_NTOK, 256, 1);  // relu->bf16
  gemm_bf16_kernel<<<dim3(256 / 64, cdiv(C_NTOK, 64)), 256, 0, stream>>>(
      ffn1_bf, 1024, w_l2, 1024, l2b, ffn2, nullptr, 256, C_NTOK, 1024, 0);
  ln_kernel<<<cdiv(C_NTOK, 8), 256, 0, stream>>>(
      tgt2, ffn2, n3g, n3b, nullptr, out, nullptr, C_NTOK);   // norm3 -> d_out
}
